// QuantumAmplitudeEmbeddingModel_52450140619044
// MI455X (gfx1250) — matrix operand order repaired and checked by reading, not compiled, not run on hardware
//
#include <hip/hip_runtime.h>
#include <hip/hip_bf16.h>
#include <math.h>

// ---------------------------------------------------------------------------
// Types for CDNA5 WMMA
// ---------------------------------------------------------------------------
typedef __attribute__((ext_vector_type(16))) __bf16 v16bf;
typedef __attribute__((ext_vector_type(8)))  float  v8f;

#define M_DIM 8192
#define K_DIM 1792
#define N_DIM 1024
#define TILE_M 128
#define TILE_N 128
#define TILE_K 32
#define LDSLD 36   // ushort row stride: 72B rows -> 8B-aligned b64 stores,
                   // 18-bank skew => conflict-free column gathers

__device__ __forceinline__ unsigned short f32_to_bf16_rne(float f) {
    unsigned int u = __float_as_uint(f);
    unsigned int r = u + 0x7FFFu + ((u >> 16) & 1u);
    return (unsigned short)(r >> 16);
}

__device__ __forceinline__ unsigned int pack_bf16x2(float lo, float hi) {
    return ((unsigned int)f32_to_bf16_rne(hi) << 16) | (unsigned int)f32_to_bf16_rne(lo);
}

union BfBits { unsigned short u; __bf16 h; };

// ---------------------------------------------------------------------------
// Kernel 1: q_in = tanh(features @ W_pre + b_pre)
// bf16 WMMA, f32 accumulate. Register-pipelined global->LDS staging.
// grid: (M/128, N/128), block: 256 threads (8 wave32); wave w owns 16M x 128N.
// ---------------------------------------------------------------------------
__global__ __launch_bounds__(256)
void gemm_pre_tanh(const float* __restrict__ A,   // features  M x K
                   const float* __restrict__ B,   // W_pre     K x N
                   const float* __restrict__ bias,// b_pre     N
                   float* __restrict__ C) {       // q_in      M x N
    __shared__ unsigned short aT[TILE_M * LDSLD]; // [m][k] bf16 bits
    __shared__ unsigned short bT[TILE_N * LDSLD]; // [n][k] bf16 bits (transposed)

    const int tid  = threadIdx.x;
    const int lane = tid & 31;
    const int wave = tid >> 5;            // 0..7 -> M sub-tile
    const int mBlk = blockIdx.x * TILE_M;
    const int nBlk = blockIdx.y * TILE_N;
    const int l15  = lane & 15;
    const int hi   = (lane >= 16);

    v8f acc[8] = {};                      // 8 x (16x16 f32) accumulators
    float4 a4[4], b4[4];                  // staging regs (next K tile in flight)

    // ---- staging: A tile as float4 rows (m = idx>>3, k4 = (idx&7)*4),
    //               B tile as float4 cols (k = idx>>5, n4 = (idx&31)*4)
    auto loadTiles = [&](int k0) {
        #pragma unroll
        for (int i = 0; i < 4; ++i) {
            int idx = i * 256 + tid;
            int m = idx >> 3, k4 = (idx & 7) << 2;
            a4[i] = *(const float4*)&A[(size_t)(mBlk + m) * K_DIM + (k0 + k4)];
        }
        #pragma unroll
        for (int i = 0; i < 4; ++i) {
            int idx = i * 256 + tid;
            int k = idx >> 5, n4 = (idx & 31) << 2;
            b4[i] = *(const float4*)&B[(size_t)(k0 + k) * N_DIM + (nBlk + n4)];
        }
    };
    auto storeTiles = [&]() {
        #pragma unroll
        for (int i = 0; i < 4; ++i) {
            int idx = i * 256 + tid;
            int m = idx >> 3, k4 = (idx & 7) << 2;
            uint2 p;
            p.x = pack_bf16x2(a4[i].x, a4[i].y);
            p.y = pack_bf16x2(a4[i].z, a4[i].w);
            *(uint2*)&aT[m * LDSLD + k4] = p;          // 8B-aligned ds_store_b64
        }
        #pragma unroll
        for (int i = 0; i < 4; ++i) {
            int idx = i * 256 + tid;
            int k = idx >> 5, n4 = (idx & 31) << 2;
            bT[(n4 + 0) * LDSLD + k] = f32_to_bf16_rne(b4[i].x);
            bT[(n4 + 1) * LDSLD + k] = f32_to_bf16_rne(b4[i].y);
            bT[(n4 + 2) * LDSLD + k] = f32_to_bf16_rne(b4[i].z);
            bT[(n4 + 3) * LDSLD + k] = f32_to_bf16_rne(b4[i].w);
        }
    };

    loadTiles(0);
    for (int k0 = 0; k0 < K_DIM; k0 += TILE_K) {
        storeTiles();
        __syncthreads();
        if (k0 + TILE_K < K_DIM) loadTiles(k0 + TILE_K);  // overlap with WMMAs

        // ---- A fragment (16x32 bf16): m = lane&15; VGPR v holds K pair at
        //      (v>=4?16:0) + (lane>=16?8:0) + 2*(v&3)
        v16bf afrag;
        {
            const unsigned short* ap = &aT[(wave * 16 + l15) * LDSLD];
            #pragma unroll
            for (int v = 0; v < 8; ++v) {
                int kb = ((v >= 4) ? 16 : 0) + (hi ? 8 : 0) + 2 * (v & 3);
                BfBits b0, b1; b0.u = ap[kb]; b1.u = ap[kb + 1];
                afrag[2 * v]     = b0.h;
                afrag[2 * v + 1] = b1.h;
            }
        }

        // ---- B fragments in groups of 4 distinct register sets, then
        //      4 back-to-back WMMAs (one dscnt wait covers the group)
        #pragma unroll
        for (int g = 0; g < 2; ++g) {
            v16bf bfr[4];
            #pragma unroll
            for (int q = 0; q < 4; ++q) {
                int nt = g * 4 + q;
                const unsigned short* bp =
                    &bT[(nt * 16 + l15) * LDSLD] + (hi ? 8 : 0);
                #pragma unroll
                for (int v = 0; v < 8; ++v) {
                    int kb = ((v >= 4) ? 16 : 0) + 2 * (v & 3);
                    BfBits b0, b1; b0.u = bp[kb]; b1.u = bp[kb + 1];
                    bfr[q][2 * v]     = b0.h;
                    bfr[q][2 * v + 1] = b1.h;
                }
            }
            #pragma unroll
            for (int q = 0; q < 4; ++q) {
                acc[g * 4 + q] = __builtin_amdgcn_wmma_f32_16x16x32_bf16(
                    false, afrag, false, bfr[q], (short)0, acc[g * 4 + q],
                    false, false);
            }
        }
        __syncthreads();
    }

    // ---- epilogue: C/D layout (lane&15 = N, VGPR r = M, +8 for lanes 16..31)
    const int row0 = mBlk + wave * 16 + (hi ? 8 : 0);
    #pragma unroll
    for (int nt = 0; nt < 8; ++nt) {
        int col = nBlk + nt * 16 + l15;
        float bv = bias[col];
        #pragma unroll
        for (int r = 0; r < 8; ++r) {
            C[(size_t)(row0 + r) * N_DIM + col] = tanhf(acc[nt][r] + bv);
        }
    }
}

// ---------------------------------------------------------------------------
// Kernel 2: quantum circuit — one block per batch row, statevector in LDS
// ---------------------------------------------------------------------------
__device__ __forceinline__ float blockReduce512(float v, float* red) {
    #pragma unroll
    for (int off = 16; off > 0; off >>= 1) v += __shfl_xor(v, off, 32);
    int w = threadIdx.x >> 5;
    if ((threadIdx.x & 31) == 0) red[w] = v;
    __syncthreads();
    float s = (threadIdx.x < 16) ? red[threadIdx.x] : 0.0f;
    if (threadIdx.x < 32) {
        #pragma unroll
        for (int off = 8; off > 0; off >>= 1) s += __shfl_xor(s, off, 32);
        if (threadIdx.x == 0) red[0] = s;
    }
    __syncthreads();
    float r = red[0];
    __syncthreads();
    return r;
}

__global__ __launch_bounds__(512)
void quantum_kernel(const float* __restrict__ qin,   // 8192 x 1024
                    const float* __restrict__ qw,    // 2 x 10 x 3
                    float* __restrict__ qout) {      // 8192 x 10
    __shared__ float sre[1024];
    __shared__ float sim[1024];
    __shared__ float red[16];
    const int t = threadIdx.x;
    const int b = blockIdx.x;
    const float* row = qin + (size_t)b * 1024;

    float x0 = row[t], x1 = row[t + 512];
    float nrm2 = blockReduce512(x0 * x0 + x1 * x1, red);
    float inv = rsqrtf(nrm2);
    sre[t] = x0 * inv;  sre[t + 512] = x1 * inv;
    sim[t] = 0.0f;      sim[t + 512] = 0.0f;
    __syncthreads();

    #pragma unroll
    for (int l = 0; l < 2; ++l) {
        const int rr = (l % 9) + 1;  // entangling range: 1 then 2
        // ---- 10 Rot(phi, theta, omega) gates ----
        for (int w = 0; w < 10; ++w) {
            float phi = qw[(l * 10 + w) * 3 + 0];
            float th  = qw[(l * 10 + w) * 3 + 1];
            float om  = qw[(l * 10 + w) * 3 + 2];
            float ch = cosf(0.5f * th), sh = sinf(0.5f * th);
            float ap = -0.5f * (phi + om), am = -0.5f * (phi - om);
            float epr = cosf(ap), epi = sinf(ap);
            float emr = cosf(am), emi = sinf(am);
            // U00 = ep*c; U01 = -conj(em)*s; U10 = em*s; U11 = conj(ep)*c
            float u00r =  epr * ch, u00i =  epi * ch;
            float u01r = -emr * sh, u01i =  emi * sh;
            float u10r =  emr * sh, u10i =  emi * sh;
            float u11r =  epr * ch, u11i = -epi * ch;

            int sb = 9 - w, st = 1 << sb;
            int i0 = ((t >> sb) << (sb + 1)) | (t & (st - 1));
            int i1 = i0 + st;
            float a0r = sre[i0], a0i = sim[i0];
            float a1r = sre[i1], a1i = sim[i1];
            float n0r = u00r * a0r - u00i * a0i + u01r * a1r - u01i * a1i;
            float n0i = u00r * a0i + u00i * a0r + u01r * a1i + u01i * a1r;
            float n1r = u10r * a0r - u10i * a0i + u11r * a1r - u11i * a1i;
            float n1i = u10r * a0i + u10i * a0r + u11r * a1i + u11i * a1r;
            sre[i0] = n0r; sim[i0] = n0i;
            sre[i1] = n1r; sim[i1] = n1i;
            __syncthreads();
        }
        // ---- 10 CNOTs: control w, target (w+rr)%10 ----
        for (int w = 0; w < 10; ++w) {
            int tg = (w + rr) % 10;
            int sb = 9 - tg, st = 1 << sb;
            int i0 = ((t >> sb) << (sb + 1)) | (t & (st - 1));
            int i1 = i0 + st;
            int cbit = 1 << (9 - w);
            if (i0 & cbit) {
                float tr = sre[i0], ti = sim[i0];
                sre[i0] = sre[i1]; sim[i0] = sim[i1];
                sre[i1] = tr;      sim[i1] = ti;
            }
            __syncthreads();
        }
    }

    // ---- <Z_w> = sum_i |s_i|^2 * (1 - 2*bit_w(i)) ----
    float p0 = sre[t] * sre[t] + sim[t] * sim[t];
    float p1 = sre[t + 512] * sre[t + 512] + sim[t + 512] * sim[t + 512];
    for (int w = 0; w < 10; ++w) {
        int bm = 1 << (9 - w);
        float z = ((t & bm) ? -p0 : p0) + (((t + 512) & bm) ? -p1 : p1);
        float zz = blockReduce512(z, red);
        if (t == 0) qout[(size_t)b * 10 + w] = zz;
    }
}

// ---------------------------------------------------------------------------
// Kernel 3: main/aux heads (K=10, bandwidth-bound)
// ---------------------------------------------------------------------------
__global__ __launch_bounds__(256)
void head_kernel(const float* __restrict__ qout,
                 const float* __restrict__ Wm, const float* __restrict__ bm,
                 const float* __restrict__ Wa, const float* __restrict__ ba,
                 float* __restrict__ out) {
    int tid = blockIdx.x * blockDim.x + threadIdx.x;
    const int TOT = M_DIM * 1000;
    if (tid >= TOT) return;
    int b = tid / 1000;
    int j = tid - b * 1000;
    const float* q = qout + (size_t)b * 10;
    if (j < 250) {
        float acc = bm[j];
        #pragma unroll
        for (int k = 0; k < 10; ++k) acc += q[k] * Wm[k * 250 + j];
        out[(size_t)b * 250 + j] = acc;
    } else {
        int jj = j - 250;
        float acc = ba[jj];
        #pragma unroll
        for (int k = 0; k < 10; ++k) acc += q[k] * Wa[k * 750 + jj];
        out[(size_t)M_DIM * 250 + (size_t)b * 750 + jj] = acc;
    }
}

// ---------------------------------------------------------------------------
extern "C" void kernel_launch(void* const* d_in, const int* in_sizes, int n_in,
                              void* d_out, int out_size, void* d_ws, size_t ws_size,
                              hipStream_t stream) {
    const float* features  = (const float*)d_in[0];
    const float* W_pre     = (const float*)d_in[1];
    const float* b_pre     = (const float*)d_in[2];
    const float* q_weights = (const float*)d_in[3];
    const float* W_main    = (const float*)d_in[4];
    const float* b_main    = (const float*)d_in[5];
    const float* W_aux     = (const float*)d_in[6];
    const float* b_aux     = (const float*)d_in[7];
    float* out  = (float*)d_out;

    float* q_in  = (float*)d_ws;                       // 8192*1024 f32
    float* q_out = q_in + (size_t)M_DIM * N_DIM;       // 8192*10  f32

    dim3 g1(M_DIM / TILE_M, N_DIM / TILE_N);           // 64 x 8
    gemm_pre_tanh<<<g1, 256, 0, stream>>>(features, W_pre, b_pre, q_in);

    quantum_kernel<<<M_DIM, 512, 0, stream>>>(q_in, q_weights, q_out);

    int tot = M_DIM * 1000;
    head_kernel<<<(tot + 255) / 256, 256, 0, stream>>>(q_out, W_main, b_main,
                                                       W_aux, b_aux, out);
}
